// MPNN_9998683865479
// MI455X (gfx1250) — matrix operand order repaired and checked by reading, not compiled, not run on hardware
//
#include <hip/hip_runtime.h>
#include <math.h>

typedef __attribute__((ext_vector_type(16))) _Float16 v16h;
typedef __attribute__((ext_vector_type(8)))  _Float16 v8h;
typedef __attribute__((ext_vector_type(8)))  float    v8f;

#define WPB 8   // waves per block (wave32 -> 256 threads)
#define TPW 8   // 16-edge tiles per wave

// ---------------- generic small kernels (f32, exact) ----------------

__global__ void k_fill(float* __restrict__ p, float v, int n) {
  int i = blockIdx.x * 256 + threadIdx.x;
  if (i < n) p[i] = v;
}

__global__ void k_deg(const long long* __restrict__ col, float* __restrict__ deg, int E) {
  int e = blockIdx.x * 256 + threadIdx.x;
  if (e < E) atomicAdd(&deg[col[e]], 1.0f);
}

__global__ void k_rsqrt(float* __restrict__ p, int n) {
  int i = blockIdx.x * 256 + threadIdx.x;
  if (i < n) p[i] = rsqrtf(p[i]);   // deg >= 1 always (self loop)
}

// per-edge symmetric norm, shared by both GCN layers
__global__ void k_norm(const long long* __restrict__ row, const long long* __restrict__ col,
                       const float* __restrict__ dinv, float* __restrict__ norm, int E) {
  int e = blockIdx.x * 256 + threadIdx.x;
  if (e < E) norm[e] = dinv[row[e]] * dinv[col[e]];
}

// Y[n,F] = X[n,K] @ W[K,F]
__global__ void k_mm(const float* __restrict__ X, const float* __restrict__ W,
                     float* __restrict__ Y, int n, int K, int F) {
  int t = blockIdx.x * 256 + threadIdx.x;
  if (t >= n * F) return;
  int i = t / F, f = t % F;
  const float* xr = X + (size_t)i * K;
  float s = 0.f;
  for (int k = 0; k < K; ++k) s = fmaf(xr[k], W[k * F + f], s);
  Y[t] = s;
}

// out[i,f] = b[f] + dinv[i]^2 * hw[i,f]   (self-loop term + bias)
__global__ void k_selfinit(const float* __restrict__ hw, const float* __restrict__ dinv,
                           const float* __restrict__ b, float* __restrict__ out, int n, int F) {
  int t = blockIdx.x * 256 + threadIdx.x;
  if (t >= n * F) return;
  int i = t / F, f = t % F;
  float d = dinv[i];
  out[t] = b[f] + d * d * hw[t];
}

// out[col[e],f] += norm[e]*hw[row[e],f]
__global__ void k_scatter(const long long* __restrict__ row, const long long* __restrict__ col,
                          const float* __restrict__ norm, const float* __restrict__ hw,
                          float* __restrict__ out, int E, int F) {
  int t = blockIdx.x * 256 + threadIdx.x;
  if (t >= E * F) return;
  int e = t / F, f = t % F;
  atomicAdd(&out[col[e] * (long long)F + f], norm[e] * hw[row[e] * (long long)F + f]);
}

__global__ void k_tanh(float* __restrict__ p, int n) {
  int i = blockIdx.x * 256 + threadIdx.x;
  if (i < n) p[i] = tanhf(p[i]);
}

// ---------------- EdgeConv: per-edge MLP on WMMA ----------------
// t = [h[col] | h[row]-h[col]]  (16 edges x 128)  f16, f32 accumulate
// m = relu(t@W1+b1)@W2+b2 ; h2[col] = max over edges (h2 pre-zeroed => fused relu)

__device__ inline void load8_f16(const float* __restrict__ hc, const float* __restrict__ hr,
                                 int k0, _Float16* dst) {
  if (k0 < 64) {
#pragma unroll
    for (int j = 0; j < 8; ++j) dst[j] = (_Float16)hc[k0 + j];
  } else {
    int k = k0 - 64;
#pragma unroll
    for (int j = 0; j < 8; ++j) dst[j] = (_Float16)(hr[k + j] - hc[k + j]);
  }
}

__global__ __launch_bounds__(256) void k_edgeconv(
    const float* __restrict__ h1,
    const long long* __restrict__ row, const long long* __restrict__ col,
    const float* __restrict__ w1, const float* __restrict__ b1,
    const float* __restrict__ w2, const float* __restrict__ b2,
    float* __restrict__ h2, int E)
{
  __shared__ __align__(32) _Float16 sW1t[64][128];    // transposed [n][k], 16 KB
  __shared__ float sB1[64], sB2[64];
  __shared__ __align__(32) _Float16 sM1[WPB][16][64]; // per-wave layer-1 out, 16 KB

  const int tid = threadIdx.x;
  for (int idx = tid; idx < 128 * 64; idx += 256) {   // w1: [k=128][n=64]
    int k = idx >> 6, n = idx & 63;
    sW1t[n][k] = (_Float16)w1[idx];
  }
  if (tid < 64) { sB1[tid] = b1[tid]; sB2[tid] = b2[tid]; }
  __syncthreads();

  const int wave   = tid >> 5;
  const int lane   = tid & 31;
  const int laneM  = lane & 15;     // A row; also B/C column (N)
  const int hiHalf = lane >> 4;     // lane group 0/1
  const int laneN  = laneM;

  // ---- layer-2 B fragments: resident in VGPRs for the whole wave ----
  // B layout: elem i -> K = ch*32 + i + (i>=8?8:0) + hiHalf*8 ; N = laneN
  v16h B2[2][4];
#pragma unroll
  for (int ch = 0; ch < 2; ++ch)
#pragma unroll
    for (int nt = 0; nt < 4; ++nt) {
      union { v16h v; _Float16 e[16]; } u;
#pragma unroll
      for (int i = 0; i < 16; ++i)
        u.e[i] = (_Float16)w2[(ch * 32 + i + (i >= 8 ? 8 : 0) + hiHalf * 8) * 64 + nt * 16 + laneN];
      B2[ch][nt] = u.v;
    }

  const int waveBase = (blockIdx.x * WPB + wave) * (TPW * 16);

#pragma unroll 1
  for (int t = 0; t < TPW; ++t) {
    const int tileBase = waveBase + t * 16;

    int eA  = tileBase + laneM;
    int eAc = (eA < E) ? eA : (E - 1);        // clamp: EXEC stays all-ones
    long long rI = row[eAc], cI = col[eAc];
    const float* hr = h1 + rI * 64;
    const float* hc = h1 + cI * 64;

    // prefetch next tile's node rows (global_prefetch_b8; rows sit in L2)
    if (t + 1 < TPW) {
      int eP  = tileBase + 16 + laneM;
      int ePc = (eP < E) ? eP : (E - 1);
      __builtin_prefetch(h1 + row[ePc] * 64, 0, 0);
      __builtin_prefetch(h1 + col[ePc] * 64, 0, 0);
    }

    // ---- layer-1 A fragments (16x128, 4 K-chunks of 32) ----
    // A layout: elem i -> K = ch*32 + i + (i>=8?8:0) + hiHalf*8
    v16h A[4];
#pragma unroll
    for (int ch = 0; ch < 4; ++ch) {
      union { v16h v; _Float16 e[16]; } u;
      int kb = ch * 32 + hiHalf * 8;
      load8_f16(hc, hr, kb,      &u.e[0]);    // i = 0..7
      load8_f16(hc, hr, kb + 16, &u.e[8]);    // i = 8..15
      A[ch] = u.v;
    }

    // ---- layer 1: m1 = relu(t @ W1 + b1) -> LDS (f16) ----
#pragma unroll
    for (int nt = 0; nt < 4; ++nt) {
      float bv = sB1[nt * 16 + laneN];
      v8f acc = { bv, bv, bv, bv, bv, bv, bv, bv };
      const _Float16* wrow = &sW1t[nt * 16 + laneN][0];
#pragma unroll
      for (int ch = 0; ch < 4; ++ch) {
        union { v16h v; v8h h[2]; } ub;
        ub.h[0] = *(const v8h*)(wrow + ch * 32 + hiHalf * 8);
        ub.h[1] = *(const v8h*)(wrow + ch * 32 + hiHalf * 8 + 16);
        v16h B = ub.v;
        acc = __builtin_amdgcn_wmma_f32_16x16x32_f16(false, A[ch], false, B,
                                                     (short)0, acc, false, false);
      }
#pragma unroll
      for (int v = 0; v < 8; ++v) {           // C layout: M = v + hiHalf*8
        float m = acc[v];
        m = m > 0.f ? m : 0.f;                // relu
        sM1[wave][v + hiHalf * 8][nt * 16 + laneN] = (_Float16)m;
      }
    }

    __syncthreads();  // uniform trip count across all waves; orders sM1 W->R

    // ---- layer-2 A fragments from LDS (16x64, 2 K-chunks) ----
    v16h A2[2];
#pragma unroll
    for (int ch = 0; ch < 2; ++ch) {
      union { v16h v; v8h h[2]; } u;
      const _Float16* mrow = &sM1[wave][laneM][0];
      u.h[0] = *(const v8h*)(mrow + ch * 32 + hiHalf * 8);       // i=0..7
      u.h[1] = *(const v8h*)(mrow + ch * 32 + hiHalf * 8 + 16);  // i=8..15
      A2[ch] = u.v;
    }

    // ---- layer 2 + fused segment_max/relu scatter ----
#pragma unroll
    for (int nt = 0; nt < 4; ++nt) {
      float bv = sB2[nt * 16 + laneN];
      v8f acc = { bv, bv, bv, bv, bv, bv, bv, bv };
#pragma unroll
      for (int ch = 0; ch < 2; ++ch)
        acc = __builtin_amdgcn_wmma_f32_16x16x32_f16(false, A2[ch], false, B2[ch][nt],
                                                     (short)0, acc, false, false);
#pragma unroll
      for (int v = 0; v < 8; ++v) {
        int e = tileBase + v + hiHalf * 8;
        float m = acc[v];
        if (e < E && m > 0.f) {               // h2 pre-zeroed: only positives matter
          long long c = col[e];
          float* dst = h2 + c * 64 + nt * 16 + laneN;
          // racy pre-filter: atomicMax is monotone, the atomic is the authority.
          if (m > *(volatile float*)dst)
            atomicMax((int*)dst, __float_as_int(m));  // int order == float order for >0
        }
      }
    }

    __syncthreads();  // keep sM1 stable before next iteration overwrites
  }
}

// ---------------- launch ----------------

static inline unsigned gdim(long long n) { return (unsigned)((n + 255) / 256); }

extern "C" void kernel_launch(void* const* d_in, const int* in_sizes, int n_in,
                              void* d_out, int out_size, void* d_ws, size_t ws_size,
                              hipStream_t stream) {
  (void)n_in; (void)out_size; (void)ws_size;
  const float*     x   = (const float*)d_in[0];
  const long long* ei  = (const long long*)d_in[1];   // int64 [2, E]
  const float*     g1w = (const float*)d_in[2];
  const float*     g1b = (const float*)d_in[3];
  const float*     ew1 = (const float*)d_in[4];
  const float*     eb1 = (const float*)d_in[5];
  const float*     ew2 = (const float*)d_in[6];
  const float*     eb2 = (const float*)d_in[7];
  const float*     g2w = (const float*)d_in[8];
  const float*     g2b = (const float*)d_in[9];
  float* out = (float*)d_out;

  const int N = in_sizes[0] / 32;
  const int E = in_sizes[1] / 2;
  const long long* row = ei;
  const long long* col = ei + E;

  float* ws   = (float*)d_ws;
  float* dinv = ws;                         // N
  float* hw1  = dinv + N;                   // N*64
  float* h1   = hw1 + (size_t)N * 64;       // N*64  (accum, then tanh in place)
  float* h2   = h1  + (size_t)N * 64;       // N*64  (edgeconv out, pre-relu'd)
  float* hw2  = h2  + (size_t)N * 64;       // N*16
  float* enorm= hw2 + (size_t)N * 16;       // E

  // symmetric normalization (shared by both GCN layers)
  k_fill <<<gdim(N), 256, 0, stream>>>(dinv, 1.0f, N);   // self loop
  k_deg  <<<gdim(E), 256, 0, stream>>>(col, dinv, E);
  k_rsqrt<<<gdim(N), 256, 0, stream>>>(dinv, N);
  k_norm <<<gdim(E), 256, 0, stream>>>(row, col, dinv, enorm, E);

  // GCN1 + tanh
  k_mm      <<<gdim((long long)N * 64), 256, 0, stream>>>(x, g1w, hw1, N, 32, 64);
  k_selfinit<<<gdim((long long)N * 64), 256, 0, stream>>>(hw1, dinv, g1b, h1, N, 64);
  k_scatter <<<gdim((long long)E * 64), 256, 0, stream>>>(row, col, enorm, hw1, h1, E, 64);
  k_tanh    <<<gdim((long long)N * 64), 256, 0, stream>>>(h1, N * 64);

  // EdgeConv (WMMA) + fused relu/segment_max
  k_fill<<<gdim((long long)N * 64), 256, 0, stream>>>(h2, 0.0f, N * 64);
  const long long edgesPerBlock = (long long)WPB * TPW * 16;
  unsigned eblocks = (unsigned)((E + edgesPerBlock - 1) / edgesPerBlock);
  k_edgeconv<<<eblocks, 256, 0, stream>>>(h1, row, col, ew1, eb1, ew2, eb2, h2, E);

  // GCN2 -> out
  k_mm      <<<gdim((long long)N * 16), 256, 0, stream>>>(h2, g2w, hw2, N, 64, 16);
  k_selfinit<<<gdim((long long)N * 16), 256, 0, stream>>>(hw2, dinv, g2b, out, N, 16);
  k_scatter <<<gdim((long long)E * 16), 256, 0, stream>>>(row, col, enorm, hw2, out, E, 16);
}
